// SlidingWindowAttention_43825846288469
// MI455X (gfx1250) — matrix operand order repaired and checked by reading, not compiled, not run on hardware
//
#include <hip/hip_runtime.h>

typedef _Float16 f16;
typedef __attribute__((ext_vector_type(8)))  _Float16 v8h;
typedef __attribute__((ext_vector_type(16))) _Float16 v16h;
typedef __attribute__((ext_vector_type(8)))  float    v8f;
typedef __attribute__((ext_vector_type(4)))  int      v4i;

#define NEG_INF (-__builtin_inff())

#ifndef __has_builtin
#define __has_builtin(x) 0
#endif
#if defined(__HIP_DEVICE_COMPILE__) && \
    __has_builtin(__builtin_amdgcn_global_load_async_to_lds_b128) && \
    __has_builtin(__builtin_amdgcn_s_wait_asynccnt)
#define HAVE_ASYNC_LDS 1
#else
#define HAVE_ASYNC_LDS 0
#endif

#if HAVE_ASYNC_LDS
typedef __attribute__((address_space(1))) v4i gas_v4i;  // global (__device__) int4
typedef __attribute__((address_space(3))) v4i las_v4i;  // LDS int4
#define ASYNC_WAIT_NEXT() __builtin_amdgcn_s_wait_asynccnt(4)
#define ASYNC_WAIT_ALL()  __builtin_amdgcn_s_wait_asynccnt(0)
#else
#define ASYNC_WAIT_NEXT()
#define ASYNC_WAIT_ALL()
#endif

// copy 16 bytes (8 f16) global -> LDS, async when available
static __device__ __forceinline__ void cp16_lds(f16* l, const f16* g) {
#if HAVE_ASYNC_LDS
  __builtin_amdgcn_global_load_async_to_lds_b128((gas_v4i*)g, (las_v4i*)l, 0, 0);
#else
  *(v8h*)l = *(const v8h*)g;
#endif
}

static __device__ __forceinline__ v8f wmma_f16(v16h a, v16h b, v8f c) {
  // D = A(16x32 f16) * B(32x16 f16) + C(16x16 f32)
  return __builtin_amdgcn_wmma_f32_16x16x32_f16(false, a, false, b, (short)0, c, false, false);
}

static __device__ __forceinline__ v16h cat8(v8h lo, v8h hi) {
  return __builtin_shufflevector(lo, hi, 0,1,2,3,4,5,6,7,8,9,10,11,12,13,14,15);
}

// A fragment (16x32) from row-major f16 matrix in global memory.
// Lane L (half=L/16, r=L%16): elem e -> K = (e&7) + ((e>>3)<<4) + 8*half.
static __device__ __forceinline__ v16h load_a_g(const f16* __restrict__ A, int lda,
                                                int row0, int k0, int lane) {
  int half = lane >> 4, r = lane & 15;
  const f16* p = A + (size_t)(row0 + r) * lda + k0 + half * 8;
  return cat8(*(const v8h*)p, *(const v8h*)(p + 16));
}

// B fragment (32x16) where B[k][n] is stored N-major (Bn[n][k] row-major NxK, global).
// Lane L: col n; elem e -> K = (e&7) + ((e>>3)<<4) + 8*half.
static __device__ __forceinline__ v16h load_b_nmajor(const f16* __restrict__ Bn, int ldb,
                                                     int k0, int n, int lane) {
  int half = lane >> 4;
  const f16* p = Bn + (size_t)n * ldb + k0 + half * 8;
  return cat8(*(const v8h*)p, *(const v8h*)(p + 16));
}

// ---------------- fp32 -> f16 convert ----------------
__global__ void cvt_kernel(const float* __restrict__ s, f16* __restrict__ d, int n) {
  int i = (blockIdx.x * blockDim.x + threadIdx.x) * 4;
  if (i < n) {
    float4 x = *(const float4*)(s + i);
    d[i + 0] = (f16)x.x; d[i + 1] = (f16)x.y;
    d[i + 2] = (f16)x.z; d[i + 3] = (f16)x.w;
  }
}

// ---------------- LDS-staged tiled GEMM: Y = X @ W^T + b ----------------
// Block: 256 threads = 8 waves; tile 128(M) x 128(N); K-step 32, double-buffered LDS.
// Wave (wm = wave&3, wn = wave>>2) computes rows wm*32..+31, cols wn*64..+63.
// MODE 0: store f16, permuted [B,H,S,dh], with scale.   MODE 1: store fp32 [M,N].
#define LSTR 40  // LDS row stride in f16 (80B: 16B-aligned, conflict-free for b128)
template <int MODE>
__global__ __launch_bounds__(256)
void linear_tile_kernel(const f16* __restrict__ X, const f16* __restrict__ Wn,
                        const float* __restrict__ bias,
                        f16* __restrict__ dstp, float* __restrict__ dstf,
                        int M, int K, int N, int S, int H, int dh, float scale) {
  __shared__ f16 xs[2][128 * LSTR];
  __shared__ f16 wsl[2][128 * LSTR];
  int tid = threadIdx.x;
  int lane = tid & 31, wave = tid >> 5;
  int half = lane >> 4, ln = lane & 15;
  int bm = blockIdx.y * 128, bn = blockIdx.x * 128;
  int wm = wave & 3, wn = wave >> 2;

  // staging: thread t loads 16 f16 of X and 16 f16 of W per K-step
  int srow = tid >> 1, sc = (tid & 1) * 16;
  const f16* gX = X + (size_t)(bm + srow) * K + sc;
  const f16* gW = Wn + (size_t)(bn + srow) * K + sc;

  auto stage = [&](int buf, int k0) {
    f16* lx = &xs[buf][srow * LSTR + sc];
    f16* lw = &wsl[buf][srow * LSTR + sc];
    cp16_lds(lx, gX + k0);     cp16_lds(lx + 8, gX + k0 + 8);
    cp16_lds(lw, gW + k0);     cp16_lds(lw + 8, gW + k0 + 8);
  };

  v8f acc[2][4] = {};
  int nk = K >> 5;
  stage(0, 0);
  for (int kk = 0; kk < nk; ++kk) {
    int buf = kk & 1;
    if (kk + 1 < nk) { stage(buf ^ 1, (kk + 1) << 5); ASYNC_WAIT_NEXT(); }
    else             { ASYNC_WAIT_ALL(); }
    __syncthreads();
    const f16* xb = &xs[buf][0];
    const f16* wb = &wsl[buf][0];
    v16h a0, a1;
    {
      const f16* p = xb + (wm * 32 + ln) * LSTR + half * 8;
      a0 = cat8(*(const v8h*)p, *(const v8h*)(p + 16));
      p += 16 * LSTR;
      a1 = cat8(*(const v8h*)p, *(const v8h*)(p + 16));
    }
#pragma unroll
    for (int t = 0; t < 4; ++t) {
      const f16* p = wb + (wn * 64 + t * 16 + ln) * LSTR + half * 8;
      v16h b = cat8(*(const v8h*)p, *(const v8h*)(p + 16));
      acc[0][t] = wmma_f16(a0, b, acc[0][t]);
      acc[1][t] = wmma_f16(a1, b, acc[1][t]);
    }
    __syncthreads();
  }

#pragma unroll
  for (int ms = 0; ms < 2; ++ms) {
#pragma unroll
    for (int t = 0; t < 4; ++t) {
      int n = bn + wn * 64 + t * 16 + ln;
      float bvv = bias[n];
#pragma unroll
      for (int vv = 0; vv < 8; ++vv) {
        int m = bm + wm * 32 + ms * 16 + vv + 8 * half;
        float y = acc[ms][t][vv] + bvv;
        if (MODE == 0) {
          int hh = n / dh, dcol = n % dh;
          int bb = m / S, s = m % S;
          dstp[(((size_t)bb * H + hh) * S + s) * dh + dcol] = (f16)(y * scale);
        } else {
          dstf[(size_t)m * N + n] = y;
        }
      }
    }
  }
}

// ---------------- sliding-window flash attention ----------------
// Q,K,V f16 [B,H,S,dh]; O f16 [B*S, H*dh]. Scale folded into Q.
#define WIN 256
#define VSTR 72  // padded V-tile stride in f16
__global__ __launch_bounds__(128)
void swa_kernel(const f16* __restrict__ Q, const f16* __restrict__ Km,
                const f16* __restrict__ V, f16* __restrict__ O,
                int S, int H, int dh) {
  __shared__ f16 psh[4][16 * 32];
  __shared__ f16 vsh[4][32 * VSTR];
  int lane = threadIdx.x & 31;
  int wave = threadIdx.x >> 5;
  int half = lane >> 4, ln = lane & 15;
  int qs = blockIdx.x * 64 + wave * 16;
  int h = blockIdx.y;
  int b = blockIdx.z;
  size_t bh = (size_t)b * H + h;
  const f16* q  = Q + bh * S * dh;
  const f16* kp = Km + bh * S * dh;
  const f16* vp = V + bh * S * dh;

  v16h aq0 = load_a_g(q, dh, qs, 0, lane);
  v16h aq1 = load_a_g(q, dh, qs, 32, lane);

  float mrow[8], lrow[8];
#pragma unroll
  for (int i = 0; i < 8; ++i) { mrow[i] = NEG_INF; lrow[i] = 0.f; }
  v8f o[4] = {};

  int jlo = qs - WIN; if (jlo < 0) jlo = 0; jlo &= ~31;
  int jhi = qs + 15 + WIN; if (jhi > S - 1) jhi = S - 1;

  f16* ps = psh[wave];
  f16* vw = vsh[wave];

  for (int ks = jlo; ks <= jhi; ks += 32) {
    // ---- stage this wave's 32x64 V tile into LDS (coalesced b128 reads) ----
    {
      int j = ks + lane; if (j > S - 1) j = S - 1;  // P==0 for OOB rows
      const f16* src = vp + (size_t)j * dh;
      f16* dst = vw + lane * VSTR;
#pragma unroll
      for (int qq = 0; qq < 8; ++qq)
        *(v8h*)(dst + qq * 8) = *(const v8h*)(src + qq * 8);
    }
    // ---- scores: 16 queries x 32 keys over K=64 ----
    v8f sc[2] = {};
#pragma unroll
    for (int nt = 0; nt < 2; ++nt) {
      int j = ks + nt * 16 + ln; if (j > S - 1) j = S - 1;
      v16h b0 = load_b_nmajor(kp, dh, 0, j, lane);
      v16h b1 = load_b_nmajor(kp, dh, 32, j, lane);
      sc[nt] = wmma_f16(aq0, b0, sc[nt]);
      sc[nt] = wmma_f16(aq1, b1, sc[nt]);
    }
    // ---- masked online softmax (fp32) ----
    float p[2][8], bm[8];
#pragma unroll
    for (int vv = 0; vv < 8; ++vv) {
      int i = qs + vv + 8 * half;
      float best = NEG_INF;
#pragma unroll
      for (int nt = 0; nt < 2; ++nt) {
        int j = ks + nt * 16 + ln;
        int d = i - j;
        bool ok = (j < S) && (d <= WIN) && (d >= -WIN);
        float sval = ok ? sc[nt][vv] : NEG_INF;
        p[nt][vv] = sval;
        best = fmaxf(best, sval);
      }
#pragma unroll
      for (int off = 1; off < 16; off <<= 1)
        best = fmaxf(best, __shfl_xor(best, off, 32));
      bm[vv] = best;
    }
#pragma unroll
    for (int vv = 0; vv < 8; ++vv) {
      float mn = fmaxf(mrow[vv], bm[vv]);
      float alpha = (mrow[vv] == NEG_INF) ? 0.f : __expf(mrow[vv] - mn);
      float rs = 0.f;
#pragma unroll
      for (int nt = 0; nt < 2; ++nt) {
        float pe = (p[nt][vv] == NEG_INF) ? 0.f : __expf(p[nt][vv] - mn);
        p[nt][vv] = pe;
        rs += pe;
      }
#pragma unroll
      for (int off = 1; off < 16; off <<= 1)
        rs += __shfl_xor(rs, off, 32);
      mrow[vv] = mn;
      lrow[vv] = lrow[vv] * alpha + rs;
#pragma unroll
      for (int t = 0; t < 4; ++t) o[t][vv] *= alpha;
    }
    // ---- transpose P through per-wave LDS into an A fragment (16x32 f16) ----
#pragma unroll
    for (int vv = 0; vv < 8; ++vv) {
      int row = vv + 8 * half;
      ps[row * 32 + ln]      = (f16)p[0][vv];
      ps[row * 32 + 16 + ln] = (f16)p[1][vv];
    }
    v16h ap = cat8(*(const v8h*)(ps + ln * 32 + half * 16),
                   *(const v8h*)(ps + ln * 32 + half * 16 + 8));
    // ---- O += P (16x32) x V (32x64), V B-fragments gathered from LDS ----
#pragma unroll
    for (int t = 0; t < 4; ++t) {
      v16h bvf;
      int n = t * 16 + ln;
#pragma unroll
      for (int e = 0; e < 16; ++e)
        bvf[e] = vw[(16 * half + e) * VSTR + n];
      o[t] = wmma_f16(ap, bvf, o[t]);
    }
  }

  // ---- normalize + store f16 [B*S, H*dh] ----
  int D = H * dh;
#pragma unroll
  for (int t = 0; t < 4; ++t) {
    int col = h * dh + t * 16 + ln;
#pragma unroll
    for (int vv = 0; vv < 8; ++vv) {
      int row = b * S + qs + vv + 8 * half;
      O[(size_t)row * D + col] = (f16)(o[t][vv] / lrow[vv]);
    }
  }
}

extern "C" void kernel_launch(void* const* d_in, const int* in_sizes, int n_in,
                              void* d_out, int out_size, void* d_ws, size_t ws_size,
                              hipStream_t stream) {
  (void)in_sizes; (void)n_in; (void)out_size;
  const float* x  = (const float*)d_in[0];
  const float* wq = (const float*)d_in[1];
  const float* bq = (const float*)d_in[2];
  const float* wk = (const float*)d_in[3];
  const float* bk = (const float*)d_in[4];
  const float* wv = (const float*)d_in[5];
  const float* bv = (const float*)d_in[6];
  const float* wo = (const float*)d_in[7];
  const float* bo = (const float*)d_in[8];
  float* out = (float*)d_out;

  const int B = 2, S = 2048, D = 1024, H = 16, dh = 64, M = B * S;
  const size_t MD = (size_t)M * D, DD = (size_t)D * D;
  if (ws_size < (MD * 5 + DD * 4) * sizeof(f16)) return;

  f16* xh  = (f16*)d_ws;
  f16* wqh = xh + MD;
  f16* wkh = wqh + DD;
  f16* wvh = wkh + DD;
  f16* woh = wvh + DD;
  f16* qh  = woh + DD;
  f16* kh  = qh + MD;
  f16* vh  = kh + MD;
  f16* ah  = vh + MD;

  int n1 = (int)MD, n2 = (int)DD;
  cvt_kernel<<<dim3((n1 / 4 + 255) / 256), dim3(256), 0, stream>>>(x, xh, n1);
  cvt_kernel<<<dim3((n2 / 4 + 255) / 256), dim3(256), 0, stream>>>(wq, wqh, n2);
  cvt_kernel<<<dim3((n2 / 4 + 255) / 256), dim3(256), 0, stream>>>(wk, wkh, n2);
  cvt_kernel<<<dim3((n2 / 4 + 255) / 256), dim3(256), 0, stream>>>(wv, wvh, n2);
  cvt_kernel<<<dim3((n2 / 4 + 255) / 256), dim3(256), 0, stream>>>(wo, woh, n2);

  dim3 gg(D / 128, M / 128);  // (8, 32)
  dim3 bb(256);
  const float scale = 0.125f;  // dh^-0.5, folded into Q
  linear_tile_kernel<0><<<gg, bb, 0, stream>>>(xh, wqh, bq, qh, nullptr, M, D, D, S, H, dh, scale);
  linear_tile_kernel<0><<<gg, bb, 0, stream>>>(xh, wkh, bk, kh, nullptr, M, D, D, S, H, dh, 1.0f);
  linear_tile_kernel<0><<<gg, bb, 0, stream>>>(xh, wvh, bv, vh, nullptr, M, D, D, S, H, dh, 1.0f);

  swa_kernel<<<dim3(S / 64, H, B), dim3(128), 0, stream>>>(qh, kh, vh, ah, S, H, dh);

  linear_tile_kernel<1><<<gg, bb, 0, stream>>>(ah, woh, bo, nullptr, out, M, D, D, S, H, dh, 1.0f);
}
